// AdaptiveuBCLLoss_49881750176192
// MI455X (gfx1250) — matrix operand order repaired and checked by reading, not compiled, not run on hardware
//
#include <hip/hip_runtime.h>
#include <hip/hip_bf16.h>

typedef __attribute__((ext_vector_type(16))) __bf16 v16bf;
typedef __attribute__((ext_vector_type(8)))  float  v8f;

union Frag32B { uint4 u[2]; v16bf v; };

#define N_ROWS 4096
#define D_DIM  1024
#define KSTEP  64
#define NSTEPS (D_DIM / KSTEP)          // 16
#define BROW_ELTS 72                    // padded LDS row stride (bf16 elts, 144 B)
#define BBUF_ELTS (128 * BROW_ELTS)     // one B buffer: 128 rows x 72 elts = 18432 B

// ---------------- block reduction helper (wave32) ----------------
__device__ __forceinline__ float block_reduce_sum(float v, volatile float* sm) {
#pragma unroll
    for (int m = 16; m; m >>= 1) v += __shfl_xor(v, m, 32);
    int lane = threadIdx.x & 31, w = threadIdx.x >> 5;
    if (lane == 0) sm[w] = v;
    __syncthreads();
    if (threadIdx.x < 32) {
        v = (lane < 8) ? sm[lane] : 0.0f;
#pragma unroll
        for (int m = 4; m; m >>= 1) v += __shfl_xor(v, m, 32);
        if (lane == 0) sm[0] = v;
    }
    __syncthreads();
    float r = sm[0];
    __syncthreads();
    return r;
}

// ---------------- kernel 1: normalize rows, cast to bf16, diag cosine ----------------
__global__ __launch_bounds__(256) void normalize_kernel(
        const float* __restrict__ out4,   // [4096][2][1024]
        unsigned short* __restrict__ z1n, // bf16 [4096][1024]
        unsigned short* __restrict__ z2n, // bf16 [4096][1024]
        float* __restrict__ diag)         // [4096] = C_ii
{
    __shared__ float sm[8];
    const int i = blockIdx.x;
    const int t = threadIdx.x;
    const float* r1 = out4 + (size_t)i * (2 * D_DIM);
    const float* r2 = r1 + D_DIM;
    float4 a = ((const float4*)r1)[t];
    float4 b = ((const float4*)r2)[t];

    float s1 = a.x*a.x + a.y*a.y + a.z*a.z + a.w*a.w;
    float s2 = b.x*b.x + b.y*b.y + b.z*b.z + b.w*b.w;
    float sd = a.x*b.x + a.y*b.y + a.z*b.z + a.w*b.w;

    s1 = block_reduce_sum(s1, sm);
    s2 = block_reduce_sum(s2, sm);
    sd = block_reduce_sum(sd, sm);

    const float n1 = fmaxf(sqrtf(s1), 1e-8f);
    const float n2 = fmaxf(sqrtf(s2), 1e-8f);
    const float i1 = 1.0f / n1, i2 = 1.0f / n2;

    union { __bf16 h[4]; uint2 u; } p1, p2;
    p1.h[0] = (__bf16)(a.x * i1); p1.h[1] = (__bf16)(a.y * i1);
    p1.h[2] = (__bf16)(a.z * i1); p1.h[3] = (__bf16)(a.w * i1);
    p2.h[0] = (__bf16)(b.x * i2); p2.h[1] = (__bf16)(b.y * i2);
    p2.h[2] = (__bf16)(b.z * i2); p2.h[3] = (__bf16)(b.w * i2);
    ((uint2*)(z1n + (size_t)i * D_DIM))[t] = p1.u;
    ((uint2*)(z2n + (size_t)i * D_DIM))[t] = p2.u;

    if (t == 0) diag[i] = sd * i1 * i2;
}

// ---------------- kernel 2: bf16 WMMA GEMM fused with exp + row partial sums ----------
// block = 256 threads = 8 waves; block tile = 128(M) x 128(N); wave tile = 16(M) x 128(N)
// B (z2n K-slice, 128 cols x 64 K = 16 KB) staged per block per K-step into LDS via
// async DMA (double-buffered, ASYNCcnt), shared by all 8 waves. A register-direct.
// grid = (32, 32): x -> M macro tile (128 rows), y -> N strip (128 cols).
__global__ __launch_bounds__(256) void gemm_lse_kernel(
        const unsigned short* __restrict__ z1u,
        const unsigned short* __restrict__ z2u,
        const float* __restrict__ lambda_,
        float* __restrict__ Spart)        // [4096][32]
{
    __shared__ __align__(16) unsigned short bsh[2][BBUF_ELTS];   // 2 x 18432 B

    const __bf16* z1n = (const __bf16*)z1u;
    const float lam = lambda_[0];

    const int tid  = threadIdx.x;
    const int lane = tid & 31;
    const int wave = tid >> 5;
    const int m0 = blockIdx.x * 128 + wave * 16;
    const int n0 = blockIdx.y * 128;
    const int l15 = lane & 15;
    const int hi  = lane >> 4;          // 0 or 1
    const int hiA = hi * 8;             // A: K sub-offset 0 / 8 (second chunk +16)
    const int hiB = hi * 16;            // unused; B also reads at hiA (0 / 8, second chunk +16)

    const __bf16* arow = z1n + (size_t)(m0 + l15) * D_DIM + hiA;

    // --- B staging: 1024 16-byte chunks per K-step (128 rows x 8 chunks),
    //     each thread moves 4 chunks. chunk index = wave*128 + j*32 + lane.
    int srow[4], sck[4];
    const unsigned short* gsrc[4];
    unsigned int ldst[4];
    const unsigned int ldsBase = (unsigned int)(uintptr_t)(&bsh[0][0]);
#pragma unroll
    for (int j = 0; j < 4; ++j) {
        const int c = wave * 128 + j * 32 + lane;
        srow[j] = c >> 3;               // 0..127
        sck[j]  = c & 7;                // 0..7 (16B chunk within 64-K row)
        gsrc[j] = z2u + (size_t)(n0 + srow[j]) * D_DIM + sck[j] * 8;
        ldst[j] = ldsBase + (unsigned int)(srow[j] * (BROW_ELTS * 2) + sck[j] * 16);
    }
    const unsigned int bufStride = (unsigned int)(BBUF_ELTS * 2);   // bytes

    v8f acc[8];
#pragma unroll
    for (int t = 0; t < 8; ++t)
        acc[t] = (v8f){0.f,0.f,0.f,0.f,0.f,0.f,0.f,0.f};

    // prologue: stage K-step 0 into buffer 0
#pragma unroll
    for (int j = 0; j < 4; ++j)
        asm volatile("global_load_async_to_lds_b128 %0, %1, off"
                     :: "v"(ldst[j]), "v"(gsrc[j]) : "memory");
    asm volatile("s_wait_asynccnt 0" ::: "memory");
    __syncthreads();

    for (int s = 0; s < NSTEPS; ++s) {
        const int k0  = s * KSTEP;
        const int cur = s & 1;

        // stage next K-step into the other buffer (overlaps with WMMA below)
        if (s + 1 < NSTEPS) {
            const unsigned int nb = (unsigned int)(1 - cur) * bufStride;
#pragma unroll
            for (int j = 0; j < 4; ++j)
                asm volatile("global_load_async_to_lds_b128 %0, %1, off"
                             :: "v"(ldst[j] + nb), "v"(gsrc[j] + k0 + KSTEP) : "memory");
        }

        __builtin_prefetch((const void*)(arow + k0 + 256), 0, 3);
        const unsigned short* bbase = &bsh[cur][0];

#pragma unroll
        for (int kk = 0; kk < KSTEP; kk += 32) {
            Frag32B a;
            a.u[0] = *(const uint4*)(arow + k0 + kk);
            a.u[1] = *(const uint4*)(arow + k0 + kk + 16);
#pragma unroll
            for (int t = 0; t < 8; ++t) {
                Frag32B b;
                const unsigned short* bp =
                    bbase + (t * 16 + l15) * BROW_ELTS + kk + hiA;
                b.u[0] = *(const uint4*)(bp);
                b.u[1] = *(const uint4*)(bp + 16);
                acc[t] = __builtin_amdgcn_wmma_f32_16x16x32_bf16(
                    false, a.v, false, b.v, (short)0, acc[t], false, false);
            }
        }

        // drain this iteration's async stage, then block-sync buffers
        asm volatile("s_wait_asynccnt 0" ::: "memory");
        __syncthreads();
    }

    // epilogue: exp + row sums, reduce across each 16-lane half
    float s[8];
#pragma unroll
    for (int r = 0; r < 8; ++r) s[r] = 0.0f;
#pragma unroll
    for (int t = 0; t < 8; ++t)
#pragma unroll
        for (int r = 0; r < 8; ++r)
            s[r] += __expf(lam * acc[t][r]);

#pragma unroll
    for (int r = 0; r < 8; ++r) {
        s[r] += __shfl_xor(s[r], 1, 32);
        s[r] += __shfl_xor(s[r], 2, 32);
        s[r] += __shfl_xor(s[r], 4, 32);
        s[r] += __shfl_xor(s[r], 8, 32);
    }
    // C layout: VGPR r -> row m0+r (lanes 0-15), row m0+8+r (lanes 16-31)
    if (l15 == 0) {
        const int rbase = m0 + hi * 8;
#pragma unroll
        for (int r = 0; r < 8; ++r)
            Spart[(size_t)(rbase + r) * 32 + blockIdx.y] = s[r];
    }
}

// ---------------- kernel 3: final reduction to scalar mean loss ----------------------
__global__ __launch_bounds__(256) void finalize_kernel(
        const float* __restrict__ Spart,
        const float* __restrict__ diag,
        const float* __restrict__ lambda_,
        float* __restrict__ out)
{
    __shared__ float sm[8];
    const float lam = lambda_[0];
    float acc = 0.0f;
    for (int i = threadIdx.x; i < N_ROWS; i += 256) {
        const float* p = Spart + (size_t)i * 32;
        float ssum = 0.0f;
#pragma unroll
        for (int k = 0; k < 32; ++k) ssum += p[k];
        acc += logf(ssum) - lam * diag[i];
    }
    acc = block_reduce_sum(acc, sm);
    if (threadIdx.x == 0) out[0] = acc * (1.0f / (float)N_ROWS);
}

extern "C" void kernel_launch(void* const* d_in, const int* in_sizes, int n_in,
                              void* d_out, int out_size, void* d_ws, size_t ws_size,
                              hipStream_t stream) {
    const float* out4    = (const float*)d_in[0];  // (4096, 2, 1024) f32
    const float* lambda_ = (const float*)d_in[1];  // (1,) f32

    char* ws = (char*)d_ws;
    const size_t Zbytes = (size_t)N_ROWS * D_DIM * 2;     // 8 MiB per bf16 matrix
    unsigned short* z1n = (unsigned short*)(ws);
    unsigned short* z2n = (unsigned short*)(ws + Zbytes);
    float* diag  = (float*)(ws + 2 * Zbytes);                          // 16 KiB
    float* Spart = (float*)(ws + 2 * Zbytes + (size_t)N_ROWS * 4);     // 512 KiB

    normalize_kernel<<<N_ROWS, 256, 0, stream>>>(out4, z1n, z2n, diag);
    gemm_lse_kernel<<<dim3(32, 32), 256, 0, stream>>>(z1n, z2n, lambda_, Spart);
    finalize_kernel<<<1, 256, 0, stream>>>(Spart, diag, lambda_, (float*)d_out);
}
